// CombinedContentLoss_19567871001233
// MI455X (gfx1250) — matrix operand order repaired and checked by reading, not compiled, not run on hardware
//
#include <hip/hip_runtime.h>
#include <math.h>

typedef __attribute__((ext_vector_type(16))) _Float16 v16h;
typedef __attribute__((ext_vector_type(8)))  _Float16 v8h;
typedef __attribute__((ext_vector_type(8)))  float    v8f;

#define BB   4
#define CC   256
#define NN   4096          // 64*64 spatial positions
#define NT   (NN/16)       // 256 tiles of 16
#define EPS1 1e-8f
#define EPS2 1e-5f
#define INV_H 1.0f
#define MSE_W 1.0f
#define PERC_W 0.02f

// workspace layout in float units
#define OFF_SCAL 0                       // [0]=mse_sum [1]=perc_sum [2..5]=cs_sum[b]
#define OFF_MX   8                       // B*C channel means of X
#define OFF_MY   (OFF_MX + BB*CC)
#define OFF_XN   (OFF_MY + BB*CC)        // B*N row norms of X
#define OFF_YN   (OFF_XN + BB*NN)
#define OFF_DMIN (OFF_YN + BB*NN)        // B*N per-row d-min
#define OFF_WSUM (OFF_DMIN + BB*NN)      // B*N per-row w-sum
#define OFF_HALF (OFF_WSUM + BB*NN)      // f16 region start (byte 270368, 32B aligned)

__device__ __forceinline__ float fast_rcp(float x) {
    return __builtin_amdgcn_rcpf(x);     // v_rcp_f32 (TRANS, co-executes with WMMA)
}

__global__ void zero_scalars_k(float* wsf) {
    if (threadIdx.x < 8) wsf[threadIdx.x] = 0.0f;
}

// grid-stride sum of squared differences, block-reduce, atomic accumulate
__global__ void sqdiff_reduce_k(const float* __restrict__ a, const float* __restrict__ b,
                                long n, float* accum) {
    __shared__ float sm[256];
    float s = 0.0f;
    for (long i = (long)blockIdx.x * blockDim.x + threadIdx.x; i < n;
         i += (long)gridDim.x * blockDim.x) {
        float d = a[i] - b[i];
        s += d * d;
    }
    sm[threadIdx.x] = s; __syncthreads();
    for (int o = 128; o > 0; o >>= 1) {
        if ((int)threadIdx.x < o) sm[threadIdx.x] += sm[threadIdx.x + o];
        __syncthreads();
    }
    if (threadIdx.x == 0) atomicAdd(accum, sm[0]);
}

// per (b,c) mean over the 4096 contiguous positions; grid.x = B*C, grid.y in {0:X, 1:Y}
__global__ void channel_means_k(const float* __restrict__ xf, const float* __restrict__ yf,
                                float* wsf) {
    const float* src = blockIdx.y ? yf : xf;
    float* dst = wsf + (blockIdx.y ? OFF_MY : OFF_MX);
    const float* p = src + (size_t)blockIdx.x * NN;
    __shared__ float sm[256];
    float s = 0.0f;
    for (int i = threadIdx.x; i < NN; i += 256) s += p[i];
    sm[threadIdx.x] = s; __syncthreads();
    for (int o = 128; o > 0; o >>= 1) {
        if ((int)threadIdx.x < o) sm[threadIdx.x] += sm[threadIdx.x + o];
        __syncthreads();
    }
    if (threadIdx.x == 0) dst[blockIdx.x] = sm[0] * (1.0f / NN);
}

// center per channel, stage as row-major [b, n, c] f16, compute row L2 norms.
// grid.x = B*(N/256), grid.y in {0:X, 1:Y}, block = 256 (thread <-> position n)
__global__ void center_norm_k(const float* __restrict__ xf, const float* __restrict__ yf,
                              float* wsf) {
    const int which = blockIdx.y;
    const float* src  = which ? yf : xf;
    const float* mean = wsf + (which ? OFF_MY : OFF_MX);
    _Float16* dst = (_Float16*)(wsf + OFF_HALF) + (which ? (size_t)BB * NN * CC : 0);
    float* nrm = wsf + (which ? OFF_YN : OFF_XN);

    const int b  = blockIdx.x / (NN / 256);
    const int n  = (blockIdx.x % (NN / 256)) * 256 + threadIdx.x;
    const float* sb = src + (size_t)b * CC * NN + n;       // coalesced reads over n
    const float* mb = mean + b * CC;                       // uniform -> scalar loads
    _Float16* db = dst + ((size_t)b * NN + n) * CC;

    float ss = 0.0f;
    for (int c0 = 0; c0 < CC; c0 += 8) {
        v8h pack;
        #pragma unroll
        for (int j = 0; j < 8; j++) {
            float v = sb[(size_t)(c0 + j) * NN] - mb[c0 + j];
            ss = fmaf(v, v, ss);
            pack[j] = (_Float16)v;
        }
        *(v8h*)(db + c0) = pack;                           // 16B store per thread
    }
    nrm[b * NN + n] = sqrtf(ss);
}

// A-fragment (16x32 f16): lane l<16 holds row M=l, K = kk*32+0..15;
// lanes 16-31 hold K = kk*32+16..31. One contiguous 32B load per lane.
__device__ __forceinline__ v16h load_a_frag(const _Float16* __restrict__ M, int row, int kbase) {
    return *(const v16h*)(M + (size_t)row * CC + kbase);
}

// B-fragment (32x16 f16, B = Yc^T): lane l<16 holds column N=l (= Yc row), K = kk*32+0..15;
// lanes 16-31 hold K = kk*32+16..31. One contiguous 32B load per lane.
__device__ __forceinline__ v16h load_b_frag(const _Float16* __restrict__ M, int row, int kbase) {
    return *(const v16h*)(M + (size_t)row * CC + kbase);
}

// Pass 1+2: one wave owns 16 x-rows, sweeps all 4096 y twice.
// Sweep 1 -> dmin[x] (row min of d = 1 - cos). Sweep 2 -> wsum[x] = sum_y exp((1-d_)/h).
__global__ void __launch_bounds__(128) ctx_pass12_k(float* wsf) {
    const _Float16* Xc = (const _Float16*)(wsf + OFF_HALF);
    const _Float16* Yc = Xc + (size_t)BB * NN * CC;
    const int wave = threadIdx.x >> 5;
    const int lane = threadIdx.x & 31;
    const int lp   = lane & 15;
    const int half = lane >> 4;
    const int gw = blockIdx.x * 4 + wave;
    const int b  = gw / NT;
    const int x0 = (gw % NT) * 16;

    const _Float16* Xb = Xc + (size_t)b * NN * CC;
    const _Float16* Yb = Yc + (size_t)b * NN * CC;
    const float* xn = wsf + OFF_XN + b * NN;
    const float* yn = wsf + OFF_YN + b * NN;

    v16h afrag[8];
    #pragma unroll
    for (int kk = 0; kk < 8; kk++)
        afrag[kk] = load_a_frag(Xb, x0 + lp, kk * 32 + half * 16);

    float xnv[8];
    #pragma unroll
    for (int i = 0; i < 8; i++) xnv[i] = xn[x0 + half * 8 + i];

    float rmax[8];
    #pragma unroll
    for (int i = 0; i < 8; i++) rmax[i] = -3.4e38f;

    for (int yt = 0; yt < NT; yt++) {
        const int y0 = yt * 16;
        v8f acc = {};
        #pragma unroll
        for (int kk = 0; kk < 8; kk++) {
            v16h bf = load_b_frag(Yb, y0 + lp, kk * 32 + half * 16);
            acc = __builtin_amdgcn_wmma_f32_16x16x32_f16(false, afrag[kk], false, bf,
                                                         (short)0, acc, false, false);
        }
        const float ynl = yn[y0 + lp];
        #pragma unroll
        for (int i = 0; i < 8; i++) {
            float r = acc[i] * fast_rcp(fmaf(xnv[i], ynl, EPS1));   // cosine
            rmax[i] = fmaxf(rmax[i], r);
        }
    }
    // row max over the 16 lanes of each half (same rows) -> dmin = 1 - max(cos)
    #pragma unroll
    for (int i = 0; i < 8; i++)
        for (int m = 1; m < 16; m <<= 1)
            rmax[i] = fmaxf(rmax[i], __shfl_xor(rmax[i], m, 32));

    float dmn[8], idm_h[8], base_h[8];
    #pragma unroll
    for (int i = 0; i < 8; i++) {
        dmn[i] = 1.0f - rmax[i];
        float idm = fast_rcp(dmn[i] + EPS2);
        // w = exp((1 - (1-r)*idm)/h) = exp(r*idm/h + (1-idm)/h)
        idm_h[i]  = idm * INV_H;
        base_h[i] = (1.0f - idm) * INV_H;
    }

    float wsm[8] = {0.f, 0.f, 0.f, 0.f, 0.f, 0.f, 0.f, 0.f};
    for (int yt = 0; yt < NT; yt++) {
        const int y0 = yt * 16;
        v8f acc = {};
        #pragma unroll
        for (int kk = 0; kk < 8; kk++) {
            v16h bf = load_b_frag(Yb, y0 + lp, kk * 32 + half * 16);
            acc = __builtin_amdgcn_wmma_f32_16x16x32_f16(false, afrag[kk], false, bf,
                                                         (short)0, acc, false, false);
        }
        const float ynl = yn[y0 + lp];
        #pragma unroll
        for (int i = 0; i < 8; i++) {
            float r = acc[i] * fast_rcp(fmaf(xnv[i], ynl, EPS1));
            wsm[i] += __expf(fmaf(r, idm_h[i], base_h[i]));
        }
    }
    #pragma unroll
    for (int i = 0; i < 8; i++)
        for (int m = 1; m < 16; m <<= 1)
            wsm[i] += __shfl_xor(wsm[i], m, 32);

    if (lp == 0) {
        #pragma unroll
        for (int i = 0; i < 8; i++) {
            const int row = x0 + half * 8 + i;
            wsf[OFF_DMIN + b * NN + row] = dmn[i];
            wsf[OFF_WSUM + b * NN + row] = wsm[i];
        }
    }
}

// Pass 3: one wave owns 16 y-columns, sweeps all x-tiles; colmax_y = max_x w/wsum[x];
// accumulate sum over y into cs_sum[b].
__global__ void __launch_bounds__(128) ctx_pass3_k(float* wsf) {
    const _Float16* Xc = (const _Float16*)(wsf + OFF_HALF);
    const _Float16* Yc = Xc + (size_t)BB * NN * CC;
    const int wave = threadIdx.x >> 5;
    const int lane = threadIdx.x & 31;
    const int lp   = lane & 15;
    const int half = lane >> 4;
    const int gw = blockIdx.x * 4 + wave;
    const int b  = gw / NT;
    const int y0 = (gw % NT) * 16;

    const _Float16* Xb = Xc + (size_t)b * NN * CC;
    const _Float16* Yb = Yc + (size_t)b * NN * CC;
    const float* xn = wsf + OFF_XN + b * NN;
    const float* yn = wsf + OFF_YN + b * NN;
    const float* dmin = wsf + OFF_DMIN + b * NN;
    const float* wsum = wsf + OFF_WSUM + b * NN;

    v16h bfrag[8];
    #pragma unroll
    for (int kk = 0; kk < 8; kk++)
        bfrag[kk] = load_b_frag(Yb, y0 + lp, kk * 32 + half * 16);
    const float ynl = yn[y0 + lp];

    float cmax = 0.0f;  // c > 0 always
    for (int xt = 0; xt < NT; xt++) {
        const int x0 = xt * 16;
        float xnv[8], idm_h[8], base_lw[8];
        #pragma unroll
        for (int i = 0; i < 8; i++) {
            const int row = x0 + half * 8 + i;
            xnv[i] = xn[row];
            float idm = fast_rcp(dmin[row] + EPS2);
            idm_h[i] = idm * INV_H;
            // fold 1/wsum into the exponent: c = exp(arg) / wsum = exp(arg - ln(wsum))
            base_lw[i] = (1.0f - idm) * INV_H - __logf(wsum[row]);
        }
        v8f acc = {};
        #pragma unroll
        for (int kk = 0; kk < 8; kk++) {
            v16h af = load_a_frag(Xb, x0 + lp, kk * 32 + half * 16);
            acc = __builtin_amdgcn_wmma_f32_16x16x32_f16(false, af, false, bfrag[kk],
                                                         (short)0, acc, false, false);
        }
        #pragma unroll
        for (int i = 0; i < 8; i++) {
            float r = acc[i] * fast_rcp(fmaf(xnv[i], ynl, EPS1));
            float c = __expf(fmaf(r, idm_h[i], base_lw[i]));
            cmax = fmaxf(cmax, c);
        }
    }
    // combine the two row-halves (rows 0-7 vs 8-15 of each tile) for the same column
    cmax = fmaxf(cmax, __shfl_xor(cmax, 16, 32));
    if (lane < 16) atomicAdd(&wsf[2 + b], cmax);
}

__global__ void finalize_k(const float* wsf, float* out) {
    float mse  = wsf[0] / (float)(4 * 3 * 256 * 256);
    float perc = wsf[1] / (float)((long)BB * CC * NN);
    float cs   = (wsf[2] + wsf[3] + wsf[4] + wsf[5]) / (float)(BB * NN);
    out[0] = MSE_W * mse - logf(cs) + PERC_W * perc;
}

extern "C" void kernel_launch(void* const* d_in, const int* in_sizes, int n_in,
                              void* d_out, int out_size, void* d_ws, size_t ws_size,
                              hipStream_t stream) {
    const float* outputs = (const float*)d_in[0];
    const float* targets = (const float*)d_in[1];
    const float* of      = (const float*)d_in[2];   // output_features [4,256,64,64]
    const float* tf      = (const float*)d_in[3];   // target_features
    float* wsf = (float*)d_ws;
    float* out = (float*)d_out;

    zero_scalars_k<<<1, 32, 0, stream>>>(wsf);
    sqdiff_reduce_k<<<256, 256, 0, stream>>>(outputs, targets, (long)4 * 3 * 256 * 256, wsf + 0);
    sqdiff_reduce_k<<<512, 256, 0, stream>>>(of, tf, (long)BB * CC * NN, wsf + 1);
    {
        dim3 g(BB * CC, 2);
        channel_means_k<<<g, 256, 0, stream>>>(of, tf, wsf);
    }
    {
        dim3 g(BB * (NN / 256), 2);
        center_norm_k<<<g, 256, 0, stream>>>(of, tf, wsf);
    }
    ctx_pass12_k<<<BB * NT / 4, 128, 0, stream>>>(wsf);
    ctx_pass3_k<<<BB * NT / 4, 128, 0, stream>>>(wsf);
    finalize_k<<<1, 1, 0, stream>>>(wsf, out);
}
